// highwayNet_1202590843338
// MI455X (gfx1250) — matrix operand order repaired and checked by reading, not compiled, not run on hardware
//
#include <hip/hip_runtime.h>
#include <hip/hip_bf16.h>

typedef __attribute__((ext_vector_type(16))) _Float16 v16h;
typedef __attribute__((ext_vector_type(8)))  float    v8f;

#define BB   2048
#define TT   16
#define KNB  12
#define GG   (TT*BB)       // 32768 groups
#define BKK  (BB*KNB)      // 24576
#define ENCH 64
#define TEF  144
#define DECH 128
#define TOUTN 25

__device__ __forceinline__ float lrelu1(float x) { return x > 0.f ? x : 0.1f*x; }
__device__ __forceinline__ float lrelu2(float x) { return x > 0.f ? x : 0.2f*x; }
__device__ __forceinline__ float sigm(float x)   { return 1.f/(1.f+__expf(-x)); }

// ---------------------------------------------------------------------------
// Fused 2-layer GAT per group (13 nodes: ego + 12 neighbors, self loops).
// Block = 128 threads (4 waves). Node-feature GEMMs via v_wmma_f32_16x16x32_f16.
// ---------------------------------------------------------------------------
__global__ void gat_group_kernel(
    const float* __restrict__ X,
    const float* __restrict__ ipW, const float* __restrict__ ipB,
    const float* __restrict__ W1,  const float* __restrict__ as1,
    const float* __restrict__ ad1, const float* __restrict__ b1,
    const float* __restrict__ W2,  const float* __restrict__ as2,
    const float* __restrict__ ad2, const float* __restrict__ b2,
    float* __restrict__ hist_gat, float* __restrict__ nbrs_gat)
{
    __shared__ float sa[16*64];
    __shared__ float sx[16*64];
    __shared__ float ss[16], sd[16], sega[16], sna[16*2];

    const int g    = blockIdx.x;
    const int tid  = threadIdx.x;
    const int lane = tid & 31;
    const int wave = tid >> 5;
    const long nodeBase = (long)g * 13;
    const int m  = lane & 15;
    const int hl = lane >> 4;

    // h0 = lrelu(X @ ipW^T + ipB), rows 13..15 zero padded (stride 32)
    for (int e = tid; e < 16*32; e += 128) {
        int r = e >> 5, j = e & 31;
        float v = 0.f;
        if (r < 13) {
            const float* xr = X + (nodeBase + r)*4;
            v = xr[0]*ipW[j*4+0] + xr[1]*ipW[j*4+1] +
                xr[2]*ipW[j*4+2] + xr[3]*ipW[j*4+3] + ipB[j];
            v = lrelu1(v);
        }
        sa[r*32 + j] = v;
    }
    __syncthreads();

    // GEMM1: sx(16x64) = sa(16x32) @ W1^T  (W1 is 64x32)
    {
        v16h a, bm; v8f acc;
        #pragma unroll
        for (int i = 0; i < 8; ++i) acc[i] = 0.f;
        #pragma unroll
        for (int v = 0; v < 8; ++v) {
            int kk = ((v>>2)<<4) + (hl<<3) + ((v&3)<<1);
            a[2*v]   = (_Float16)sa[m*32 + kk];
            a[2*v+1] = (_Float16)sa[m*32 + kk + 1];
        }
        int n = wave*16 + m;
        #pragma unroll
        for (int v = 0; v < 8; ++v) {
            int k = ((v>>2)<<4) + (hl<<3) + ((v&3)<<1);
            bm[2*v]   = (_Float16)W1[n*32 + k];
            bm[2*v+1] = (_Float16)W1[n*32 + k + 1];
        }
        acc = __builtin_amdgcn_wmma_f32_16x16x32_f16(false, a, false, bm,
                                                     (short)0, acc, false, false);
        #pragma unroll
        for (int v = 0; v < 8; ++v) sx[(v + (hl<<3))*64 + n] = acc[v];
    }
    __syncthreads();

    // attention scalars s = xw@a_s, d = xw@a_d
    if (tid < 26) {
        int r = tid % 13;
        const float* row = sx + r*64;
        const float* av  = (tid < 13) ? as1 : ad1;
        float acc = 0.f;
        for (int k = 0; k < 64; ++k) acc += row[k]*av[k];
        if (tid < 13) ss[r] = acc; else sd[r] = acc;
    }
    __syncthreads();

    // softmax alphas (leaky 0.2). ego dst: 13 sources; nbr dst: {ego,self}
    if (tid == 0) {
        float mx = -1e30f;
        for (int s = 0; s < 13; ++s) { float t = lrelu2(ss[s]+sd[0]); sega[s]=t; mx=fmaxf(mx,t); }
        float den = 0.f;
        for (int s = 0; s < 13; ++s) { float v = __expf(sega[s]-mx); sega[s]=v; den+=v; }
        for (int s = 0; s < 13; ++s) sega[s] /= den;
    } else if (tid < 13) {
        int r = tid;
        float e0 = lrelu2(ss[0]+sd[r]);
        float e1 = lrelu2(ss[r]+sd[r]);
        float mx = fmaxf(e0,e1);
        float x0 = __expf(e0-mx), x1 = __expf(e1-mx), den = x0+x1;
        sna[r*2+0] = x0/den; sna[r*2+1] = x1/den;
    }
    __syncthreads();

    // aggregate -> h1 = lrelu(agg + b1) into sa (stride 64), pad rows
    for (int e = tid; e < 16*64; e += 128) {
        int r = e >> 6, c = e & 63;
        float v = 0.f;
        if (r == 0) {
            float a = 0.f;
            for (int s = 0; s < 13; ++s) a += sega[s]*sx[s*64 + c];
            v = lrelu1(a + b1[c]);
        } else if (r < 13) {
            float a = sna[r*2+0]*sx[c] + sna[r*2+1]*sx[r*64 + c];
            v = lrelu1(a + b1[c]);
        }
        sa[r*64 + c] = v;
    }
    __syncthreads();

    // GEMM2: sx(16x32) = sa(16x64) @ W2^T  (W2 is 32x64), waves 0..1
    if (wave < 2) {
        v8f acc;
        #pragma unroll
        for (int i = 0; i < 8; ++i) acc[i] = 0.f;
        int n = wave*16 + m;
        #pragma unroll
        for (int kc = 0; kc < 64; kc += 32) {
            v16h a, bm;
            #pragma unroll
            for (int v = 0; v < 8; ++v) {
                int kk = kc + ((v>>2)<<4) + (hl<<3) + ((v&3)<<1);
                a[2*v]   = (_Float16)sa[m*64 + kk];
                a[2*v+1] = (_Float16)sa[m*64 + kk + 1];
            }
            #pragma unroll
            for (int v = 0; v < 8; ++v) {
                int k = kc + ((v>>2)<<4) + (hl<<3) + ((v&3)<<1);
                bm[2*v]   = (_Float16)W2[n*64 + k];
                bm[2*v+1] = (_Float16)W2[n*64 + k + 1];
            }
            acc = __builtin_amdgcn_wmma_f32_16x16x32_f16(false, a, false, bm,
                                                         (short)0, acc, false, false);
        }
        #pragma unroll
        for (int v = 0; v < 8; ++v) sx[(v + (hl<<3))*32 + n] = acc[v];
    }
    __syncthreads();

    // second attention (GOUT=32)
    if (tid < 26) {
        int r = tid % 13;
        const float* row = sx + r*32;
        const float* av  = (tid < 13) ? as2 : ad2;
        float acc = 0.f;
        for (int k = 0; k < 32; ++k) acc += row[k]*av[k];
        if (tid < 13) ss[r] = acc; else sd[r] = acc;
    }
    __syncthreads();
    if (tid == 0) {
        float mx = -1e30f;
        for (int s = 0; s < 13; ++s) { float t = lrelu2(ss[s]+sd[0]); sega[s]=t; mx=fmaxf(mx,t); }
        float den = 0.f;
        for (int s = 0; s < 13; ++s) { float v = __expf(sega[s]-mx); sega[s]=v; den+=v; }
        for (int s = 0; s < 13; ++s) sega[s] /= den;
    } else if (tid < 13) {
        int r = tid;
        float e0 = lrelu2(ss[0]+sd[r]);
        float e1 = lrelu2(ss[r]+sd[r]);
        float mx = fmaxf(e0,e1);
        float x0 = __expf(e0-mx), x1 = __expf(e1-mx), den = x0+x1;
        sna[r*2+0] = x0/den; sna[r*2+1] = x1/den;
    }
    __syncthreads();

    // aggregate + b2, lrelu, write outputs directly
    for (int e = tid; e < 13*32; e += 128) {
        int r = e >> 5, c = e & 31;
        float v;
        if (r == 0) {
            float a = 0.f;
            for (int s = 0; s < 13; ++s) a += sega[s]*sx[s*32 + c];
            v = lrelu1(a + b2[c]);
            hist_gat[(long)g*32 + c] = v;
        } else {
            float a = sna[r*2+0]*sx[c] + sna[r*2+1]*sx[r*32 + c];
            v = lrelu1(a + b2[c]);
            nbrs_gat[((long)g*KNB + (r-1))*32 + c] = v;
        }
    }
}

// ---------------------------------------------------------------------------
// Feature concat: [pos(32) | va(32) | dis(16) | lane(32) | gat(32)] = 144
// ---------------------------------------------------------------------------
__global__ void build_feat_kernel(
    const float* __restrict__ xp, const float* __restrict__ xv,
    const float* __restrict__ xd, const float* __restrict__ xl,
    const float* __restrict__ gat,
    const float* __restrict__ posW, const float* __restrict__ posB,
    const float* __restrict__ vaW,  const float* __restrict__ vaB,
    const float* __restrict__ disW, const float* __restrict__ disB,
    const float* __restrict__ laneW,const float* __restrict__ laneB,
    float* __restrict__ out, int R)
{
    long idx = (long)blockIdx.x*blockDim.x + threadIdx.x;
    if (idx >= (long)R*TEF) return;
    int r = (int)(idx / TEF), c = (int)(idx % TEF);
    float v;
    if (c < 32) {
        v = lrelu1(xp[r*2]*posW[c*2] + xp[r*2+1]*posW[c*2+1] + posB[c]);
    } else if (c < 64) {
        int j = c-32;
        v = lrelu1(xv[r*2]*vaW[j*2] + xv[r*2+1]*vaW[j*2+1] + vaB[j]);
    } else if (c < 80) {
        int j = c-64;
        v = lrelu1(xd[r]*disW[j] + disB[j]);
    } else if (c < 112) {
        int j = c-80;
        v = lrelu1(xl[r*2]*laneW[j*2] + xl[r*2+1]*laneW[j*2+1] + laneB[j]);
    } else {
        v = gat[(long)r*32 + (c-112)];
    }
    out[idx] = v;
}

// ---------------------------------------------------------------------------
// Fused WMMA LSTM step: G = Xt@Wih^T + h@Whh^T (+ biases), gate math in-LDS.
// Block: 16 rows x 4H cols, 512 threads = 16 waves, TPW = 4H/256 tiles/wave.
// ---------------------------------------------------------------------------
template<int TPW>
__device__ __forceinline__ void mma_phase(
    const float* __restrict__ A, int K, const float* __restrict__ W,
    _Float16* As, v8f* acc, int m0, int tid, int lane, int wave)
{
    const int m = lane & 15, hl = lane >> 4;
    for (int kc = 0; kc < K; kc += 32) {
        __syncthreads();
        {
            int e = tid;                 // 512 threads, 512 elements
            int r = e >> 5, c = e & 31;
            int k = kc + c;
            As[e] = (_Float16)((k < K) ? A[(long)(m0 + r)*K + k] : 0.f);
        }
        __syncthreads();
        v16h a;
        #pragma unroll
        for (int v = 0; v < 8; ++v) {
            int kk = ((v>>2)<<4) + (hl<<3) + ((v&3)<<1);
            a[2*v]   = As[m*32 + kk];
            a[2*v+1] = As[m*32 + kk+1];
        }
        #pragma unroll
        for (int t = 0; t < TPW; ++t) {
            int n = (wave + t*16)*16 + m;
            v16h bm;
            #pragma unroll
            for (int v = 0; v < 8; ++v) {
                int k = kc + ((v>>2)<<4) + (hl<<3) + ((v&3)<<1);
                bm[2*v]   = (_Float16)((k   < K) ? W[(long)n*K + k]   : 0.f);
                bm[2*v+1] = (_Float16)((k+1 < K) ? W[(long)n*K + k+1] : 0.f);
            }
            acc[t] = __builtin_amdgcn_wmma_f32_16x16x32_f16(false, a, false, bm,
                                                            (short)0, acc[t], false, false);
        }
    }
}

template<int H>
__global__ void lstm_step_kernel(
    const float* __restrict__ Xt, int Kx, const float* __restrict__ Wih,
    const float* __restrict__ h_in, const float* __restrict__ Whh,
    const float* __restrict__ bih,  const float* __restrict__ bhh,
    float* __restrict__ c_state, float* __restrict__ h_out)
{
    constexpr int NG  = 4*H;
    constexpr int TPW = NG/256;
    __shared__ _Float16 As[16*32];
    __shared__ float    Gs[16*NG];

    const int tid  = threadIdx.x;          // 512
    const int lane = tid & 31;
    const int wave = tid >> 5;
    const int m0   = blockIdx.x * 16;
    const int m    = lane & 15;
    const int hl   = lane >> 4;

    v8f acc[TPW];
    #pragma unroll
    for (int t = 0; t < TPW; ++t)
        #pragma unroll
        for (int i = 0; i < 8; ++i) acc[t][i] = 0.f;

    mma_phase<TPW>(Xt,   Kx, Wih, As, acc, m0, tid, lane, wave);
    mma_phase<TPW>(h_in, H,  Whh, As, acc, m0, tid, lane, wave);

    #pragma unroll
    for (int t = 0; t < TPW; ++t) {
        int n = (wave + t*16)*16 + m;
        #pragma unroll
        for (int v = 0; v < 8; ++v) Gs[(v + (hl<<3))*NG + n] = acc[t][v];
    }
    __syncthreads();

    for (int e = tid; e < 16*H; e += 512) {
        int r = e / H, j = e % H;
        long mr = (long)(m0 + r);
        float gi = Gs[r*NG + j]       + bih[j]       + bhh[j];
        float gf = Gs[r*NG + H + j]   + bih[H+j]     + bhh[H+j];
        float gg = Gs[r*NG + 2*H + j] + bih[2*H+j]   + bhh[2*H+j];
        float go = Gs[r*NG + 3*H + j] + bih[3*H+j]   + bhh[3*H+j];
        float c  = sigm(gf)*c_state[mr*H + j] + sigm(gi)*tanhf(gg);
        c_state[mr*H + j] = c;
        h_out[mr*H + j]   = sigm(go)*tanhf(c);
    }
}

// ---------------------------------------------------------------------------
// Temporal attention + dyn projection
// ---------------------------------------------------------------------------
__global__ void ta_logits_kernel(const float* __restrict__ hs,
                                 const float* __restrict__ W,
                                 const float* __restrict__ bias,
                                 float* __restrict__ beta)
{
    int idx = blockIdx.x*blockDim.x + threadIdx.x;
    if (idx >= BB*TT) return;
    int b = idx >> 4, j = idx & 15;
    float acc = bias[j];
    for (int t = 0; t < TT; ++t)
        for (int h = 0; h < ENCH; ++h)
            acc += hs[((long)t*BB + b)*ENCH + h] * W[j*(TT*ENCH) + t*ENCH + h];
    beta[idx] = fmaxf(acc, 0.f);
}

__global__ void softmax16_kernel(float* __restrict__ beta)
{
    int b = blockIdx.x*blockDim.x + threadIdx.x;
    if (b >= BB) return;
    float* p = beta + b*16;
    float mx = p[0];
    for (int k = 1; k < 16; ++k) mx = fmaxf(mx, p[k]);
    float s = 0.f;
    for (int k = 0; k < 16; ++k) { float v = __expf(p[k]-mx); p[k] = v; s += v; }
    for (int k = 0; k < 16; ++k) p[k] /= s;
}

__global__ void ctx_kernel(const float* __restrict__ hs, const float* __restrict__ beta,
                           float* __restrict__ ctx)
{
    int idx = blockIdx.x*blockDim.x + threadIdx.x;
    if (idx >= BB*ENCH) return;
    int b = idx / ENCH, h = idx % ENCH;
    float a = 0.f;
    for (int t = 0; t < TT; ++t)
        a += beta[b*16 + t] * hs[((long)t*BB + b)*ENCH + h];
    ctx[idx] = a;
}

__global__ void histenc_kernel(const float* __restrict__ ctx,
                               const float* __restrict__ dynW, const float* __restrict__ dynB,
                               float* __restrict__ he)
{
    int idx = blockIdx.x*blockDim.x + threadIdx.x;
    if (idx >= BB*32) return;
    int b = idx >> 5, j = idx & 31;
    float a = dynB[j];
    for (int h = 0; h < ENCH; ++h) a += ctx[b*ENCH + h]*dynW[j*ENCH + h];
    he[idx] = lrelu1(a);
}

// ---------------------------------------------------------------------------
// Social tensor, convolutions, pooling
// ---------------------------------------------------------------------------
__global__ void soc_build_kernel(const float* __restrict__ nbrs_enc, float* __restrict__ soc)
{
    long idx = (long)blockIdx.x*blockDim.x + threadIdx.x;
    if (idx >= (long)BB*64*13*3) return;
    int b   = (int)(idx / (64*39));
    int rem = (int)(idx % (64*39));
    int c   = rem / 39;
    int pos = rem % 39;
    int x = pos / 3, y = pos % 3;
    float v = 0.f;
    if (y == 0 && x < 12) v = nbrs_enc[((long)b*KNB + x)*ENCH + c];
    soc[(((long)b*64 + c)*13 + x)*3 + y] = v;
}

__global__ void sconv_kernel(const float* __restrict__ soc, const float* __restrict__ W,
                             const float* __restrict__ bias, float* __restrict__ out)
{
    long idx = (long)blockIdx.x*blockDim.x + threadIdx.x;
    if (idx >= (long)BB*64*11) return;
    int b = (int)(idx / (64*11));
    int o = (int)((idx / 11) % 64);
    int x = (int)(idx % 11);
    float a = bias[o];
    for (int i = 0; i < 64; ++i)
        for (int kh = 0; kh < 3; ++kh)
            for (int kw = 0; kw < 3; ++kw)
                a += soc[(((long)b*64 + i)*13 + x + kh)*3 + kw] *
                     W[((o*64 + i)*3 + kh)*3 + kw];
    out[idx] = lrelu1(a);
}

__global__ void conv31_kernel(const float* __restrict__ in, const float* __restrict__ W,
                              const float* __restrict__ bias, float* __restrict__ out)
{
    long idx = (long)blockIdx.x*blockDim.x + threadIdx.x;
    if (idx >= (long)BB*16*9) return;
    int b = (int)(idx / (16*9));
    int o = (int)((idx / 9) % 16);
    int x = (int)(idx % 9);
    float a = bias[o];
    for (int i = 0; i < 64; ++i)
        for (int kh = 0; kh < 3; ++kh)
            a += in[((long)b*64 + i)*11 + x + kh] * W[(o*64 + i)*3 + kh];
    out[idx] = lrelu1(a);
}

__global__ void pool_enc_kernel(const float* __restrict__ conv2, const float* __restrict__ he,
                                float* __restrict__ enc)
{
    int idx = blockIdx.x*blockDim.x + threadIdx.x;
    if (idx >= BB*192) return;
    int b = idx / 192, q = idx % 192;
    float v;
    if (q < 160) {
        int q2 = (q < 80) ? q : q - 80;
        int c = q2 / 5, j = q2 % 5;
        int h0 = 2*j - 1, h1 = 2*j;       // padded window, pad=1
        const float* p = conv2 + ((long)b*16 + c)*9;
        if (q < 80) {
            float mx = -1e30f;
            if (h0 >= 0 && h0 < 9) mx = fmaxf(mx, p[h0]);
            if (h1 >= 0 && h1 < 9) mx = fmaxf(mx, p[h1]);
            v = mx;
        } else {
            float s = 0.f;
            if (h0 >= 0 && h0 < 9) s += p[h0];
            if (h1 >= 0 && h1 < 9) s += p[h1];
            v = s * 0.5f;
        }
    } else {
        v = he[b*32 + (q - 160)];
    }
    enc[idx] = v;
}

__global__ void latlon_kernel(const float* __restrict__ enc,
                              const float* __restrict__ latW, const float* __restrict__ latB,
                              const float* __restrict__ lonW, const float* __restrict__ lonB,
                              float* __restrict__ outLat, float* __restrict__ outLon)
{
    int b = blockIdx.x*blockDim.x + threadIdx.x;
    if (b >= BB) return;
    const float* e = enc + (long)b*192;
    float l[3];
    for (int k = 0; k < 3; ++k) {
        float a = latB[k];
        for (int q = 0; q < 192; ++q) a += e[q]*latW[k*192 + q];
        l[k] = a;
    }
    float mx = fmaxf(l[0], fmaxf(l[1], l[2]));
    float s = 0.f;
    for (int k = 0; k < 3; ++k) { l[k] = __expf(l[k]-mx); s += l[k]; }
    for (int k = 0; k < 3; ++k) outLat[b*3 + k] = l[k]/s;

    float m0 = lonB[0], m1 = lonB[1];
    for (int q = 0; q < 192; ++q) { m0 += e[q]*lonW[q]; m1 += e[q]*lonW[192 + q]; }
    float mm = fmaxf(m0, m1);
    float x0 = __expf(m0-mm), x1 = __expf(m1-mm), den = x0+x1;
    outLon[b*2 + 0] = x0/den;
    outLon[b*2 + 1] = x1/den;
}

__global__ void encd_build_kernel(const float* __restrict__ enc,
                                  const float* __restrict__ lat_enc, const float* __restrict__ lon_enc,
                                  float* __restrict__ encd)
{
    int idx = blockIdx.x*blockDim.x + threadIdx.x;
    if (idx >= BB*197) return;
    int b = idx / 197, q = idx % 197;
    float v;
    if (q < 192)      v = enc[(long)b*192 + q];
    else if (q < 195) v = lat_enc[b*3 + (q-192)];
    else              v = lon_enc[b*2 + (q-195)];
    encd[idx] = v;
}

__global__ void out_final_kernel(const float* __restrict__ hs2,
                                 const float* __restrict__ opW, const float* __restrict__ opB,
                                 float* __restrict__ out)
{
    long idx = (long)blockIdx.x*blockDim.x + threadIdx.x;
    if (idx >= (long)TOUTN*BB*5) return;
    int t = (int)(idx / (BB*5));
    int b = (int)((idx / 5) % BB);
    int q = (int)(idx % 5);
    float a = opB[q];
    const float* h = hs2 + ((long)t*BB + b)*DECH;
    for (int k = 0; k < DECH; ++k) a += h[k]*opW[q*DECH + k];
    if (q == 2 || q == 3) a = __expf(a);
    else if (q == 4)      a = tanhf(a);
    out[idx] = a;
}

// ---------------------------------------------------------------------------
static inline unsigned gdiv(long n, int b) { return (unsigned)((n + b - 1)/b); }

extern "C" void kernel_launch(void* const* d_in, const int* in_sizes, int n_in,
                              void* d_out, int out_size, void* d_ws, size_t ws_size,
                              hipStream_t stream)
{
    const float* Xfig   = (const float*)d_in[0];
    const float* hist   = (const float*)d_in[3];
    const float* histva = (const float*)d_in[4];
    const float* histdis= (const float*)d_in[5];
    const float* histln = (const float*)d_in[6];
    const float* nbrs   = (const float*)d_in[7];
    const float* nbrsva = (const float*)d_in[8];
    const float* nbrsdis= (const float*)d_in[9];
    const float* nbrsln = (const float*)d_in[10];
    const float* lat_enc= (const float*)d_in[12];
    const float* lon_enc= (const float*)d_in[13];
    const float* ipW  = (const float*)d_in[14];
    const float* ipB  = (const float*)d_in[15];
    const float* g1W  = (const float*)d_in[16];
    const float* g1as = (const float*)d_in[17];
    const float* g1ad = (const float*)d_in[18];
    const float* g1b  = (const float*)d_in[19];
    const float* g2W  = (const float*)d_in[20];
    const float* g2as = (const float*)d_in[21];
    const float* g2ad = (const float*)d_in[22];
    const float* g2b  = (const float*)d_in[23];
    const float* posW = (const float*)d_in[24];
    const float* posB = (const float*)d_in[25];
    const float* vaW  = (const float*)d_in[26];
    const float* vaB  = (const float*)d_in[27];
    const float* disW = (const float*)d_in[28];
    const float* disB = (const float*)d_in[29];
    const float* laneW= (const float*)d_in[30];
    const float* laneB= (const float*)d_in[31];
    const float* encWih = (const float*)d_in[32];
    const float* encWhh = (const float*)d_in[33];
    const float* encBih = (const float*)d_in[34];
    const float* encBhh = (const float*)d_in[35];
    const float* taWih  = (const float*)d_in[36];
    const float* taWhh  = (const float*)d_in[37];
    const float* taBih  = (const float*)d_in[38];
    const float* taBhh  = (const float*)d_in[39];
    const float* taTAW  = (const float*)d_in[40];
    const float* taTAb  = (const float*)d_in[41];
    const float* dynW   = (const float*)d_in[42];
    const float* dynB   = (const float*)d_in[43];
    const float* sconvW = (const float*)d_in[44];
    const float* sconvB = (const float*)d_in[45];
    const float* c31W   = (const float*)d_in[46];
    const float* c31B   = (const float*)d_in[47];
    const float* decWih = (const float*)d_in[48];
    const float* decWhh = (const float*)d_in[49];
    const float* decBih = (const float*)d_in[50];
    const float* decBhh = (const float*)d_in[51];
    const float* opW    = (const float*)d_in[52];
    const float* opB    = (const float*)d_in[53];
    const float* latW   = (const float*)d_in[54];
    const float* latB   = (const float*)d_in[55];
    const float* lonW   = (const float*)d_in[56];
    const float* lonB   = (const float*)d_in[57];

    float* ws = (float*)d_ws;
    size_t off = 0;
    auto take = [&](size_t n) { float* p = ws + off; off += n; return p; };

    float* hist_gat = take((size_t)GG*32);          // (T,B,32)
    float* nbrs_gat = take((size_t)GG*KNB*32);      // (T,B*K,32)
    float* ch       = take((size_t)TT*BB*TEF);      // (T,B,144)
    float* hs       = take((size_t)TT*BB*ENCH);     // (T,B,64)
    float* c_ta     = take((size_t)BB*ENCH);
    float* zeroh    = take((size_t)BKK*ENCH);       // covers B*128 too
    float* cn_t     = take((size_t)BKK*TEF);
    float* h_enc    = take((size_t)BKK*ENCH);       // final = nbrs_enc
    float* c_enc    = take((size_t)BKK*ENCH);
    float* beta     = take((size_t)BB*16);
    float* ctx      = take((size_t)BB*ENCH);
    float* he       = take((size_t)BB*32);
    float* soc      = take((size_t)BB*64*13*3);
    float* conv1    = take((size_t)BB*64*11);
    float* conv2    = take((size_t)BB*16*9);
    float* encb     = take((size_t)BB*192);
    float* encd     = take((size_t)BB*197);
    float* hs2      = take((size_t)TOUTN*BB*DECH);
    float* c_dec    = take((size_t)BB*DECH);
    (void)ws_size; (void)in_sizes; (void)n_in; (void)out_size;

    // zero the state buffers (d_ws is poisoned; keep launch deterministic)
    hipMemsetAsync(zeroh, 0, (size_t)BKK*ENCH*sizeof(float), stream);
    hipMemsetAsync(c_ta,  0, (size_t)BB*ENCH*sizeof(float), stream);
    hipMemsetAsync(c_enc, 0, (size_t)BKK*ENCH*sizeof(float), stream);
    hipMemsetAsync(c_dec, 0, (size_t)BB*DECH*sizeof(float), stream);

    // 1) fused 2-layer GAT per group
    gat_group_kernel<<<GG, 128, 0, stream>>>(Xfig, ipW, ipB,
        g1W, g1as, g1ad, g1b, g2W, g2as, g2ad, g2b, hist_gat, nbrs_gat);

    // 2) ego feature sequence ch (all T at once; hist arrays contiguous)
    build_feat_kernel<<<gdiv((long)TT*BB*TEF, 256), 256, 0, stream>>>(
        hist, histva, histdis, histln, hist_gat,
        posW, posB, vaW, vaB, disW, disB, laneW, laneB, ch, TT*BB);

    // 3) TA-LSTM over T=16 steps (store all h into hs)
    for (int t = 0; t < TT; ++t) {
        const float* hin = (t == 0) ? zeroh : (hs + (size_t)(t-1)*BB*ENCH);
        lstm_step_kernel<ENCH><<<BB/16, 512, 0, stream>>>(
            ch + (size_t)t*BB*TEF, TEF, taWih, hin, taWhh, taBih, taBhh,
            c_ta, hs + (size_t)t*BB*ENCH);
    }
    ta_logits_kernel<<<gdiv(BB*16, 256), 256, 0, stream>>>(hs, taTAW, taTAb, beta);
    softmax16_kernel<<<gdiv(BB, 256), 256, 0, stream>>>(beta);
    ctx_kernel<<<gdiv(BB*ENCH, 256), 256, 0, stream>>>(hs, beta, ctx);
    histenc_kernel<<<gdiv(BB*32, 256), 256, 0, stream>>>(ctx, dynW, dynB, he);

    // 4) neighbor encoder LSTM (per-step feature build, in-place h/c)
    for (int t = 0; t < TT; ++t) {
        build_feat_kernel<<<gdiv((long)BKK*TEF, 256), 256, 0, stream>>>(
            nbrs + (size_t)t*BKK*2, nbrsva + (size_t)t*BKK*2,
            nbrsdis + (size_t)t*BKK, nbrsln + (size_t)t*BKK*2,
            nbrs_gat + (size_t)t*BKK*32,
            posW, posB, vaW, vaB, disW, disB, laneW, laneB, cn_t, BKK);
        const float* hin = (t == 0) ? zeroh : h_enc;
        lstm_step_kernel<ENCH><<<BKK/16, 512, 0, stream>>>(
            cn_t, TEF, encWih, hin, encWhh, encBih, encBhh, c_enc, h_enc);
    }

    // 5) social tensor + convs + pooling
    soc_build_kernel<<<gdiv((long)BB*64*39, 256), 256, 0, stream>>>(h_enc, soc);
    sconv_kernel<<<gdiv((long)BB*64*11, 256), 256, 0, stream>>>(soc, sconvW, sconvB, conv1);
    conv31_kernel<<<gdiv((long)BB*16*9, 256), 256, 0, stream>>>(conv1, c31W, c31B, conv2);
    pool_enc_kernel<<<gdiv(BB*192, 256), 256, 0, stream>>>(conv2, he, encb);

    // 6) maneuver heads -> d_out tails; build decoder input encd
    float* out  = (float*)d_out;
    float* oLat = out + (size_t)TOUTN*BB*5;
    float* oLon = oLat + (size_t)BB*3;
    latlon_kernel<<<gdiv(BB, 128), 128, 0, stream>>>(encb, latW, latB, lonW, lonB, oLat, oLon);
    encd_build_kernel<<<gdiv(BB*197, 256), 256, 0, stream>>>(encb, lat_enc, lon_enc, encd);

    // 7) decoder LSTM, 25 steps, constant input encd; store all h into hs2
    for (int t = 0; t < TOUTN; ++t) {
        const float* hin = (t == 0) ? zeroh : (hs2 + (size_t)(t-1)*BB*DECH);
        lstm_step_kernel<DECH><<<BB/16, 512, 0, stream>>>(
            encd, 197, decWih, hin, decWhh, decBih, decBhh,
            c_dec, hs2 + (size_t)t*BB*DECH);
    }

    // 8) output head + transforms
    out_final_kernel<<<gdiv((long)TOUTN*BB*5, 256), 256, 0, stream>>>(hs2, opW, opB, out);
}
